// CausalSelfAttention_23235773071871
// MI455X (gfx1250) — matrix operand order repaired and checked by reading, not compiled, not run on hardware
//
#include <hip/hip_runtime.h>
#include <hip/hip_bf16.h>

#define BB 4
#define TT 2048
#define CC 1024
#define HH 16
#define DD 64

#if defined(__has_builtin)
#if __has_builtin(__builtin_amdgcn_sched_barrier)
#define SCHED_FENCE() __builtin_amdgcn_sched_barrier(0)
#endif
#endif
#ifndef SCHED_FENCE
#define SCHED_FENCE()
#endif

typedef __attribute__((ext_vector_type(16))) __bf16 v16bf;
typedef __attribute__((ext_vector_type(8)))  float  v8f;

union Frag16 { unsigned int u[8]; v16bf v; };

__device__ __forceinline__ unsigned short f2bf(float f) {
  unsigned int u = __float_as_uint(f);
  u += 0x7FFFu + ((u >> 16) & 1u);          // round-to-nearest-even
  return (unsigned short)(u >> 16);
}
__device__ __forceinline__ unsigned int pk2(unsigned short a, unsigned short b) {
  return (unsigned int)a | ((unsigned int)b << 16);
}

// ---------------------------------------------------------------------------
// Tiled GEMM with bf16 WMMA and register-double-buffered global staging.
// MODE 0: A=f32 x, epilogue scatters q/k/vT bf16 (q pre-scaled by 1/sqrt(D)).
// MODE 1: A=bf16 y, epilogue writes f32 out.
// Block tile 128x128, K-step 32, 8 waves (2x4), each wave 64x32 (4x2 frags).
// ---------------------------------------------------------------------------
template <int MODE>
__global__ __launch_bounds__(256) void gemm_bf16(
    const void* __restrict__ Ain, const float* __restrict__ Bin,
    const float* __restrict__ bias,
    unsigned short* __restrict__ qws, unsigned short* __restrict__ kws,
    unsigned short* __restrict__ vtws, float* __restrict__ out,
    int Mdim, int Ndim, int Kdim)
{
  __shared__ __align__(16) unsigned short Als[128 * 32];   // [row][k]
  __shared__ __align__(16) unsigned short Bls[128 * 32];   // [n][k] (transposed)

  const int tid   = threadIdx.x;
  const int lane  = tid & 31;
  const int wid   = tid >> 5;
  const int kh    = lane >> 4;      // lane half (0/1)
  const int lm    = lane & 15;
  const int waveM = wid >> 2;       // 0..1
  const int waveN = wid & 3;        // 0..3
  const int m0    = blockIdx.x * 128;
  const int n0    = blockIdx.y * 128;
  const int bn    = tid & 127;      // B-staging column
  const int bkg   = tid >> 7;       // B-staging k-half

  v8f acc[4][2];
  for (int i = 0; i < 4; ++i)
    for (int j = 0; j < 2; ++j)
      for (int g = 0; g < 8; ++g) acc[i][j][g] = 0.0f;

  const unsigned int* Alsu = (const unsigned int*)Als;
  const unsigned int* Blsu = (const unsigned int*)Bls;

  float4 aF[4];            // MODE 0 staging regs
  uint2  aH[4];            // MODE 1 staging regs
  float  bF[4][4];

  auto loadTile = [&](int kb) {
    if (MODE == 0) {
      const float* A = (const float*)Ain;
      for (int g = 0; g < 4; ++g) {
        int slot = tid + g * 256;               // 1024 float4 slots
        int row = slot >> 3, kc = (slot & 7) * 4;
        aF[g] = *(const float4*)(A + (size_t)(m0 + row) * Kdim + kb + kc);
      }
    } else {
      const unsigned short* A = (const unsigned short*)Ain;
      for (int g = 0; g < 4; ++g) {
        int slot = tid + g * 256;
        int row = slot >> 3, kc = (slot & 7) * 4;
        aH[g] = *(const uint2*)(A + (size_t)(m0 + row) * Kdim + kb + kc);
      }
    }
    for (int g = 0; g < 4; ++g) {
      int k0 = bkg * 16 + g * 4;
      for (int i = 0; i < 4; ++i)
        bF[g][i] = Bin[(size_t)(kb + k0 + i) * Ndim + n0 + bn];
    }
  };
  auto storeTile = [&]() {
    for (int g = 0; g < 4; ++g) {
      int slot = tid + g * 256;
      int row = slot >> 3, kc = (slot & 7) * 4;
      uint2 p;
      if (MODE == 0) {
        p.x = pk2(f2bf(aF[g].x), f2bf(aF[g].y));
        p.y = pk2(f2bf(aF[g].z), f2bf(aF[g].w));
      } else {
        p = aH[g];
      }
      ((uint2*)Als)[row * 8 + (kc >> 2)] = p;
    }
    for (int g = 0; g < 4; ++g) {
      int k0 = bkg * 16 + g * 4;
      uint2 p;
      p.x = pk2(f2bf(bF[g][0]), f2bf(bF[g][1]));
      p.y = pk2(f2bf(bF[g][2]), f2bf(bF[g][3]));
      ((uint2*)Bls)[bn * 8 + (k0 >> 2)] = p;
    }
  };

  loadTile(0);
  for (int kb = 0; kb < Kdim; kb += 32) {
    storeTile();
    __syncthreads();
    if (kb + 32 < Kdim) loadTile(kb + 32);   // overlap next tile with compute

    // ---- fragment gather (contiguous k = 16*kh + s in A and B), grouped, then 8 WMMAs
    Frag16 afr[4], bfr[2];
    for (int mi = 0; mi < 4; ++mi) {
      int row = waveM * 64 + mi * 16 + lm;
      for (int v = 0; v < 8; ++v) {
        int k = kh * 16 + v * 2;
        afr[mi].u[v] = Alsu[row * 16 + (k >> 1)];
      }
    }
    for (int nj = 0; nj < 2; ++nj) {
      int n = waveN * 32 + nj * 16 + lm;
      for (int v = 0; v < 8; ++v)
        bfr[nj].u[v] = Blsu[n * 16 + kh * 8 + v];
    }
    SCHED_FENCE();                           // keep all ds_loads above the WMMAs
    for (int mi = 0; mi < 4; ++mi)
      for (int nj = 0; nj < 2; ++nj)
        acc[mi][nj] = __builtin_amdgcn_wmma_f32_16x16x32_bf16(
            false, afr[mi].v, false, bfr[nj].v, (short)0, acc[mi][nj], false, false);
    __syncthreads();
  }

  // ---- epilogue
  for (int mi = 0; mi < 4; ++mi) {
    for (int nj = 0; nj < 2; ++nj) {
      for (int g = 0; g < 8; ++g) {
        int row = m0 + waveM * 64 + mi * 16 + g + kh * 8;
        int col = n0 + waveN * 32 + nj * 16 + lm;
        float val = acc[mi][nj][g] + bias[col];
        if (MODE == 0) {
          int which = col >> 10;                 // 0=q 1=k 2=v
          int c = col & 1023;
          int h = c >> 6, d = c & 63;
          int b = row >> 11, t = row & 2047;
          size_t hidx = ((size_t)(b * HH + h) * TT + t) * DD + d;
          if (which == 0)      qws[hidx] = f2bf(val * 0.125f);  // fold 1/sqrt(D)
          else if (which == 1) kws[hidx] = f2bf(val);
          else                 vtws[((size_t)(b * HH + h) * DD + d) * TT + t] = f2bf(val);
        } else {
          out[(size_t)row * Ndim + col] = val;
        }
      }
    }
  }
}

// ---------------------------------------------------------------------------
// Flash attention, cooperative: 8 waves / block cover 128 query rows of one
// (b,h).  K/V chunks (32 kv x 64 d) staged once per block into LDS with b128
// loads (register-double-buffered); each wave owns a 16-row strip and a
// private LDS region for the P-matrix relayout.  All WMMAs consume LDS.
// Causal mask applied only on boundary chunks (wave-uniform branch); the
// 1/sqrt(D) scale is pre-folded into Q.
// ---------------------------------------------------------------------------
__global__ __launch_bounds__(256) void attn_flash(
    const unsigned short* __restrict__ Q, const unsigned short* __restrict__ K,
    const unsigned short* __restrict__ VT, unsigned short* __restrict__ Y)
{
  __shared__ __align__(16) unsigned short Kls[32 * 64];    // [kv][d]
  __shared__ __align__(16) unsigned short Vls[64 * 32];    // [d][kv]
  __shared__ __align__(16) unsigned short Pls[8][16 * 32]; // per-wave P tile

  const int tid  = threadIdx.x;
  const int lane = tid & 31;
  const int wid  = tid >> 5;
  const int kh   = lane >> 4, lm = lane & 15;
  const int bh   = blockIdx.x >> 4;            // (b*H+h), 0..63
  const int qt   = blockIdx.x & 15;            // 16 q-tiles of 128 rows
  const int qb0  = qt * 128;
  const int qbase = qb0 + wid * 16;            // this wave's 16-row strip
  const int b = bh >> 4, h = bh & 15;

  const int krow = tid >> 3, kcol = (tid & 7) * 8;   // K stage: 32 x 64
  const int vrow = tid >> 2, vcol = (tid & 3) * 8;   // V stage: 64 x 32

  // Q A-fragments: two D halves of 32 (persist whole kernel)
  Frag16 qf[2];
  for (int dh = 0; dh < 2; ++dh) {
    const unsigned short* qrow = Q + ((size_t)bh * TT + qbase + lm) * DD + dh * 32;
    for (int v = 0; v < 8; ++v) {
      int k = kh * 16 + v * 2;
      qf[dh].u[v] = *(const unsigned int*)(qrow + k);
    }
  }

  v8f o[4];
  for (int nf = 0; nf < 4; ++nf)
    for (int g = 0; g < 8; ++g) o[nf][g] = 0.0f;
  float mrow[8], lrow[8];
  for (int g = 0; g < 8; ++g) { mrow[g] = -3.0e38f; lrow[g] = 0.0f; }

  uint4 kreg, vreg;
  auto loadKV = [&](int kv) {
    kreg = *(const uint4*)(K  + ((size_t)bh * TT + kv + krow) * DD + kcol);
    vreg = *(const uint4*)(VT + ((size_t)bh * DD + vrow) * TT + kv + vcol);
  };
  loadKV(0);

  const int jmaxB = (qb0 + 127) >> 5;          // block-wide chunk count
  const int jmaxW = (qbase + 15) >> 5;         // this wave's causal limit
  unsigned short* pw = Pls[wid];

  for (int j = 0; j <= jmaxB; ++j) {
    const int kv = j * 32;
    *(uint4*)(Kls + krow * 64 + kcol) = kreg;
    *(uint4*)(Vls + vrow * 32 + vcol) = vreg;
    __syncthreads();
    if (j < jmaxB) loadKV(kv + 32);            // overlap next chunk with compute

    if (j <= jmaxW) {
      // ---- S = Q * K^T : gather all 4 B-fragments, then 4 WMMAs
      Frag16 kf[2][2];
      for (int cf = 0; cf < 2; ++cf)
        for (int dh = 0; dh < 2; ++dh)
          for (int v = 0; v < 8; ++v)
            kf[cf][dh].u[v] = *(const unsigned int*)(
                Kls + (cf * 16 + lm) * 64 + dh * 32 + kh * 16 + 2 * v);
      SCHED_FENCE();                           // all ds_loads in flight first
      v8f s[2];
      for (int cf = 0; cf < 2; ++cf)
        for (int g = 0; g < 8; ++g) s[cf][g] = 0.0f;
      for (int cf = 0; cf < 2; ++cf)
        for (int dh = 0; dh < 2; ++dh)
          s[cf] = __builtin_amdgcn_wmma_f32_16x16x32_bf16(
              false, qf[dh].v, false, kf[cf][dh].v, (short)0, s[cf], false, false);

      // ---- causal mask: only boundary chunks need it (wave-uniform branch)
      if (kv + 31 > qbase) {
        for (int cf = 0; cf < 2; ++cf)
          for (int g = 0; g < 8; ++g) {
            int r = qbase + g + kh * 8;
            int c = kv + cf * 16 + lm;
            if (c > r) s[cf][g] = -3.0e38f;
          }
      }

      // ---- online softmax: row stats via width-16 shuffles
      for (int g = 0; g < 8; ++g) {
        float rm = fmaxf(s[0][g], s[1][g]);
        for (int off = 8; off >= 1; off >>= 1)
          rm = fmaxf(rm, __shfl_xor(rm, off, 16));
        float mnew  = fmaxf(mrow[g], rm);
        float alpha = __expf(mrow[g] - mnew);
        float p0 = __expf(s[0][g] - mnew);
        float p1 = __expf(s[1][g] - mnew);
        float rs = p0 + p1;
        for (int off = 8; off >= 1; off >>= 1)
          rs += __shfl_xor(rs, off, 16);
        lrow[g] = lrow[g] * alpha + rs;
        mrow[g] = mnew;
        for (int nf = 0; nf < 4; ++nf) o[nf][g] *= alpha;
        int rloc = g + kh * 8;
        pw[rloc * 32 + lm]      = f2bf(p0);
        pw[rloc * 32 + 16 + lm] = f2bf(p1);
      }

      // ---- O += P * V : gather P A-frag + all 4 V B-frags, then 4 WMMAs
      Frag16 pf;
      for (int v = 0; v < 8; ++v) {
        int k = kh * 16 + v * 2;
        pf.u[v] = *(const unsigned int*)(pw + lm * 32 + k);
      }
      Frag16 vf[4];
      for (int nf = 0; nf < 4; ++nf)
        for (int v = 0; v < 8; ++v)
          vf[nf].u[v] = *(const unsigned int*)(
              Vls + (nf * 16 + lm) * 32 + kh * 16 + 2 * v);
      SCHED_FENCE();                           // all ds_loads in flight first
      for (int nf = 0; nf < 4; ++nf)
        o[nf] = __builtin_amdgcn_wmma_f32_16x16x32_bf16(
            false, pf.v, false, vf[nf].v, (short)0, o[nf], false, false);
    }
    __syncthreads();
  }

  // ---- normalize, write Y [B,T,C] bf16 (token-major for the proj GEMM)
  for (int g = 0; g < 8; ++g) {
    float inv = 1.0f / lrow[g];
    int r = qbase + g + kh * 8;
    for (int nf = 0; nf < 4; ++nf) {
      int c = h * DD + nf * 16 + lm;
      Y[((size_t)b * TT + r) * CC + c] = f2bf(o[nf][g] * inv);
    }
  }
}

// ---------------------------------------------------------------------------
extern "C" void kernel_launch(void* const* d_in, const int* in_sizes, int n_in,
                              void* d_out, int out_size, void* d_ws, size_t ws_size,
                              hipStream_t stream) {
  (void)in_sizes; (void)n_in; (void)out_size; (void)ws_size;
  const float* x      = (const float*)d_in[0];
  const float* w_attn = (const float*)d_in[1];
  const float* b_attn = (const float*)d_in[2];
  const float* w_proj = (const float*)d_in[3];
  const float* b_proj = (const float*)d_in[4];
  float* out = (float*)d_out;

  const size_t per = (size_t)BB * HH * TT * DD;   // 8,388,608 elems (bf16)
  unsigned short* qws  = (unsigned short*)d_ws;
  unsigned short* kws  = qws + per;
  unsigned short* vtws = kws + per;
  unsigned short* yws  = vtws + per;              // 4 x 16.8 MB = 67 MB

  dim3 g1(8192 / 128, 3072 / 128);                // QKV GEMM
  gemm_bf16<0><<<g1, 256, 0, stream>>>(x, w_attn, b_attn,
                                       qws, kws, vtws, nullptr,
                                       8192, 3072, 1024);

  attn_flash<<<BB * HH * (TT / 128), 256, 0, stream>>>(qws, kws, vtws, yws);

  dim3 g2(8192 / 128, 1024 / 128);                // output projection
  gemm_bf16<1><<<g2, 256, 0, stream>>>(yws, w_proj, b_proj,
                                       nullptr, nullptr, nullptr, out,
                                       8192, 1024, 1024);
}
